// DurationEncoderConformer_8040178778669
// MI455X (gfx1250) — matrix operand order repaired and checked by reading, not compiled, not run on hardware
//
#include <hip/hip_runtime.h>
#include <hip/hip_bf16.h>
#include <math.h>

// ---------------------------------------------------------------------------
// Problem constants (from reference)
// ---------------------------------------------------------------------------
constexpr int LAYERS = 4;
constexpr int D      = 512;
constexpr int STY    = 128;
constexpr int H      = 8;
constexpr int T      = 512;
constexpr int BQ     = 16;      // batch
constexpr int KC     = 15;      // depthwise conv kernel
constexpr int DH     = 64;      // head dim
constexpr int FF     = 2048;
constexpr int BT     = BQ * T;  // 8192 rows
constexpr long long BTD = (long long)BT * D;

typedef __attribute__((ext_vector_type(16))) __bf16 v16bf;
typedef __attribute__((ext_vector_type(8)))  float  v8f;

#define DEV __device__ __forceinline__

DEV __bf16 to_bf16(float f) {
    unsigned u = __builtin_bit_cast(unsigned, f);
    unsigned r = u + 0x7fffu + ((u >> 16) & 1u);     // round-to-nearest-even
    unsigned short hs = (unsigned short)(r >> 16);
    return __builtin_bit_cast(__bf16, hs);
}
DEV float wredsum(float v) { for (int o = 16; o; o >>= 1) v += __shfl_xor(v, o, 32); return v; }
DEV float wredmax(float v) { for (int o = 16; o; o >>= 1) v = fmaxf(v, __shfl_xor(v, o, 32)); return v; }
DEV float siluf(float x)   { return x / (1.f + __expf(-x)); }

// ---------------------------------------------------------------------------
// WMMA GEMM:  C[M,N] = Res + outScale * act( A[M,K] * op(W) + bias )
//   op(W) = W[N,K] row-major (transB=0)  or  W^T with W[K-major] (transB=1)
// Block: 256 threads (8 waves), 128x64 tile, 32x32 per wave (4 WMMAs/K-step),
// K-step 32, bf16 WMMA, f32 accumulate.
// Requires M%128==0, N%64==0, K%32==0 (true for all uses here).
// ---------------------------------------------------------------------------
__global__ __launch_bounds__(256) void gemm_bf16(
    const float* __restrict__ A,  int lda, long long sA,
    const float* __restrict__ Wt, int ldw, long long sW, int transB,
    const float* __restrict__ bias,
    const float* __restrict__ Res,
    float* __restrict__ Cp, int ldc, long long sC,
    int Mdim, int Ndim, int Kdim, int act, float outScale)
{
    __shared__ __bf16 As[128][40];  // 32 + 8 pad
    __shared__ __bf16 Bs[64][40];

    const int z = blockIdx.z;
    A  += (long long)z * sA;
    Wt += (long long)z * sW;
    Cp += (long long)z * sC;
    if (Res) Res += (long long)z * sC;

    const int bm = blockIdx.y * 128, bn = blockIdx.x * 64;
    const int tid   = threadIdx.x;
    const int lrowA = tid >> 1;            // 0..127, 16 cols per thread
    const int lcolA = (tid & 1) * 16;
    const int lrowB = tid >> 2;            // 0..63, 8 cols per thread
    const int lcolB = (tid & 3) * 8;
    const int wid   = tid >> 5, lane = tid & 31;
    const int wr    = wid & 3;             // M sub-tile (4 x 32 rows)
    const int wc    = wid >> 2;            // N sub-tile (2 x 32 cols)
    const int hl    = lane >> 4;           // lane half 0/1
    const int lr    = lane & 15;

    v8f acc00 = {}, acc01 = {}, acc10 = {}, acc11 = {};

    for (int kk = 0; kk < Kdim; kk += 32) {
        // ---- stage A tile (128 rows x 32 K) fp32 -> bf16 ----
        {
            const float* ap = A + (long long)(bm + lrowA) * lda + kk + lcolA;
            #pragma unroll
            for (int i = 0; i < 16; i++) As[lrowA][lcolA + i] = to_bf16(ap[i]);
            if (kk + 32 < Kdim) __builtin_prefetch(ap + 32, 0, 3);  // global_prefetch_b8
        }
        // ---- stage B tile (64 N x 32 K) ----
        if (!transB) {
            const float* wp = Wt + (long long)(bn + lrowB) * ldw + kk + lcolB;
            #pragma unroll
            for (int i = 0; i < 8; i++) Bs[lrowB][lcolB + i] = to_bf16(wp[i]);
            if (kk + 32 < Kdim) __builtin_prefetch(wp + 32, 0, 3);
        } else {
            #pragma unroll
            for (int i = 0; i < 8; i++)
                Bs[lrowB][lcolB + i] = to_bf16(Wt[(long long)(kk + lcolB + i) * ldw + bn + lrowB]);
        }
        __syncthreads();

        // ---- per-lane fragments per ISA 7.12.2 (wave32 layouts) ----
        v16bf a0, a1, b0, b1;
        const __bf16* ap0 = &As[wr * 32 + lr][hl * 8];
        const __bf16* ap1 = &As[wr * 32 + 16 + lr][hl * 8];
        #pragma unroll
        for (int e = 0; e < 8; e++) {
            a0[e] = ap0[e]; a0[e + 8] = ap0[e + 16];
            a1[e] = ap1[e]; a1[e + 8] = ap1[e + 16];
        }
        const __bf16* bp0 = &Bs[wc * 32 + lr][hl * 8];
        const __bf16* bp1 = &Bs[wc * 32 + 16 + lr][hl * 8];
        #pragma unroll
        for (int e = 0; e < 8; e++) {
            b0[e] = bp0[e]; b0[e + 8] = bp0[e + 16];
            b1[e] = bp1[e]; b1[e + 8] = bp1[e + 16];
        }

        acc00 = __builtin_amdgcn_wmma_f32_16x16x32_bf16(false, a0, false, b0, (short)0, acc00, false, false);
        acc01 = __builtin_amdgcn_wmma_f32_16x16x32_bf16(false, a0, false, b1, (short)0, acc01, false, false);
        acc10 = __builtin_amdgcn_wmma_f32_16x16x32_bf16(false, a1, false, b0, (short)0, acc10, false, false);
        acc11 = __builtin_amdgcn_wmma_f32_16x16x32_bf16(false, a1, false, b1, (short)0, acc11, false, false);
        __syncthreads();
    }

    // ---- epilogue: C/D layout — lane writes rows (hl*8 + r), col lane%16 ----
    const int gn0 = bn + wc * 32 + lr;
    #pragma unroll
    for (int r = 0; r < 8; r++) {
        const int gm0 = bm + wr * 32 + hl * 8 + r;       // a0 rows
        const int gm1 = gm0 + 16;                        // a1 rows
        float v00 = acc00[r], v01 = acc01[r], v10 = acc10[r], v11 = acc11[r];
        if (bias) {
            float bz0 = bias[gn0], bz1 = bias[gn0 + 16];
            v00 += bz0; v01 += bz1; v10 += bz0; v11 += bz1;
        }
        if (act == 1) { v00 = siluf(v00); v01 = siluf(v01); v10 = siluf(v10); v11 = siluf(v11); }
        v00 *= outScale; v01 *= outScale; v10 *= outScale; v11 *= outScale;
        float* cr0 = Cp + (long long)gm0 * ldc;
        float* cr1 = Cp + (long long)gm1 * ldc;
        if (Res) {
            const float* rr0 = Res + (long long)gm0 * ldc;
            const float* rr1 = Res + (long long)gm1 * ldc;
            v00 += rr0[gn0]; v01 += rr0[gn0 + 16];
            v10 += rr1[gn0]; v11 += rr1[gn0 + 16];
        }
        cr0[gn0] = v00; cr0[gn0 + 16] = v01;
        cr1[gn0] = v10; cr1[gn0 + 16] = v11;
    }
    (void)Mdim; (void)Ndim;
}

// ---------------------------------------------------------------------------
// h[b,t,d] = x[b,d,t] + sinusoidal_pos(t,d)
// ---------------------------------------------------------------------------
__global__ void pos_embed(const float* __restrict__ x, float* __restrict__ h) {
    long long i = (long long)blockIdx.x * 256 + threadIdx.x;
    if (i >= BTD) return;
    int d = (int)(i % D);
    long long bt = i / D;
    int t = (int)(bt % T), b = (int)(bt / T);
    float div = __expf((float)(d >> 1) * (-logf(10000.f) / (float)D));
    float ang = (float)t * div;
    float pe = (d & 1) ? cosf(ang) : sinf(ang);
    h[i] = x[((long long)b * D + d) * T + t] + pe;
}

// ---------------------------------------------------------------------------
// gb[b, 0:2D] = style[b] @ aff_w^T + aff_b      (tiny matvec)
// ---------------------------------------------------------------------------
__global__ void style_affine(const float* __restrict__ style, const float* __restrict__ w,
                             const float* __restrict__ b2, float* __restrict__ gb) {
    int i = blockIdx.x * 256 + threadIdx.x;
    if (i >= BQ * 2 * D) return;
    int o = i % (2 * D), b = i / (2 * D);
    const float* sr = style + b * STY;
    const float* wr = w + (long long)o * STY;
    float acc = b2[o];
    #pragma unroll 4
    for (int k = 0; k < STY; k++) acc += sr[k] * wr[k];
    gb[i] = acc;
}

// ---------------------------------------------------------------------------
// LayerNorm over D, optional style-affine (SALN) and optional padding mask.
// One wave per row (wave32), 16 elems/lane, shuffle reductions.
// ---------------------------------------------------------------------------
__global__ __launch_bounds__(256) void saln_ln(
    const float* __restrict__ in, float* __restrict__ out,
    const float* __restrict__ lnw, const float* __restrict__ lnb,
    const float* __restrict__ gb, const unsigned char* __restrict__ mask)
{
    const int row  = blockIdx.x * 8 + (threadIdx.x >> 5);
    const int lane = threadIdx.x & 31;
    const float* xr = in + (long long)row * D;
    float v[16]; float s = 0.f;
    #pragma unroll
    for (int j = 0; j < 16; j++) { v[j] = xr[lane + j * 32]; s += v[j]; }
    s = wredsum(s);
    const float mu = s * (1.f / (float)D);
    float ss = 0.f;
    #pragma unroll
    for (int j = 0; j < 16; j++) { float dd = v[j] - mu; ss += dd * dd; }
    ss = wredsum(ss);
    const float inv = rsqrtf(ss * (1.f / (float)D) + 1e-5f);
    const int b = row / T;
    const float msc = (mask && mask[row]) ? 0.f : 1.f;
    float* orow = out + (long long)row * D;
    #pragma unroll
    for (int j = 0; j < 16; j++) {
        int d = lane + j * 32;
        float yn = (v[j] - mu) * inv * lnw[d] + lnb[d];
        if (gb) yn = (1.f + gb[b * 2 * D + d]) * yn + gb[b * 2 * D + D + d];
        orow[d] = yn * msc;
    }
}

// ---------------------------------------------------------------------------
// T5 relative-position buckets (NB=16, max_distance=64) -> tbl[T*T]
// ---------------------------------------------------------------------------
__global__ void relbucket(int* __restrict__ tbl) {
    int idx = blockIdx.x * 256 + threadIdx.x;
    if (idx >= T * T) return;
    int t = idx / T, s = idx % T;
    int rp = s - t;
    int n = rp < 0 ? -rp : rp;
    int bucket = (rp > 0) ? 8 : 0;
    int add;
    if (n < 4) add = n;
    else {
        int val = 4 + (int)(logf((float)n * 0.25f) / logf(16.f) * 4.f);
        add = val < 7 ? val : 7;
    }
    int bb = bucket + add;
    tbl[idx] = bb < 0 ? 0 : (bb > 15 ? 15 : bb);
}

// ---------------------------------------------------------------------------
// scores row: += rel_bias[tbl[t,s], h]; mask -> -1e9; softmax over s (len T)
// scores is [H,T,T] for one batch element; mrow = mask + b*T
// ---------------------------------------------------------------------------
__global__ __launch_bounds__(256) void attn_softmax(
    float* __restrict__ scores, const float* __restrict__ relb,
    const int* __restrict__ tbl, const unsigned char* __restrict__ mrow)
{
    const int t = blockIdx.x, hh = blockIdx.y, tid = threadIdx.x;
    float* row = scores + ((long long)hh * T + t) * T;
    const int s0 = tid, s1 = tid + 256;
    float v0 = row[s0] + relb[tbl[t * T + s0] * H + hh];
    float v1 = row[s1] + relb[tbl[t * T + s1] * H + hh];
    if (mrow[s0]) v0 = -1e9f;
    if (mrow[s1]) v1 = -1e9f;

    __shared__ float red[8];
    __shared__ float bcast;
    float mx = wredmax(fmaxf(v0, v1));
    if ((tid & 31) == 0) red[tid >> 5] = mx;
    __syncthreads();
    if (tid == 0) { float m = red[0]; for (int j = 1; j < 8; j++) m = fmaxf(m, red[j]); bcast = m; }
    __syncthreads();
    mx = bcast;
    v0 = __expf(v0 - mx); v1 = __expf(v1 - mx);
    float sm = wredsum(v0 + v1);
    __syncthreads();
    if ((tid & 31) == 0) red[tid >> 5] = sm;
    __syncthreads();
    if (tid == 0) { float a = 0.f; for (int j = 0; j < 8; j++) a += red[j]; bcast = a; }
    __syncthreads();
    float inv = 1.f / bcast;
    row[s0] = v0 * inv; row[s1] = v1 * inv;
}

// ---------------------------------------------------------------------------
// GLU over channel dim: out[bt,c] = hid[bt,c] * sigmoid(hid[bt,D+c])
// ---------------------------------------------------------------------------
__global__ void glu_kernel(const float* __restrict__ hid, float* __restrict__ out) {
    long long i = (long long)blockIdx.x * 256 + threadIdx.x;
    if (i >= BTD) return;
    long long bt = i / D; int c = (int)(i % D);
    float a = hid[bt * (2 * D) + c];
    float g = hid[bt * (2 * D) + D + c];
    out[i] = a * (1.f / (1.f + __expf(-g)));
}

// ---------------------------------------------------------------------------
// Depthwise conv along T (K=15, zero pad 7), [B,T,C] layout, coalesced over C.
// ---------------------------------------------------------------------------
__global__ __launch_bounds__(256) void dwconv(
    const float* __restrict__ in, float* __restrict__ out,
    const float* __restrict__ w, const float* __restrict__ bb)
{
    const int t = blockIdx.x, b = blockIdx.y;
    for (int c = threadIdx.x; c < D; c += 256) {
        float acc = bb[c];
        #pragma unroll
        for (int k = 0; k < KC; k++) {
            int tt = t + k - KC / 2;
            if (tt >= 0 && tt < T) acc += w[c * KC + k] * in[((long long)b * T + tt) * D + c];
        }
        out[((long long)b * T + t) * D + c] = acc;
    }
}

// ---------------------------------------------------------------------------
// GroupNorm(1 group) over (T,C) per batch, then *gn_w+gn_b, then SiLU. In-place.
// ---------------------------------------------------------------------------
__global__ __launch_bounds__(256) void groupnorm_silu(
    float* __restrict__ p, const float* __restrict__ gw, const float* __restrict__ gbb)
{
    const long long n = (long long)T * D;
    float* base = p + (long long)blockIdx.x * n;
    float s = 0.f, ss = 0.f;
    for (long long i = threadIdx.x; i < n; i += 256) { float v = base[i]; s += v; ss += v * v; }
    __shared__ float rs[8], rq[8];
    __shared__ float mean_s, inv_s;
    s = wredsum(s); ss = wredsum(ss);
    if ((threadIdx.x & 31) == 0) { rs[threadIdx.x >> 5] = s; rq[threadIdx.x >> 5] = ss; }
    __syncthreads();
    if (threadIdx.x == 0) {
        float a = 0.f, b2 = 0.f;
        for (int j = 0; j < 8; j++) { a += rs[j]; b2 += rq[j]; }
        float mu = a / (float)n;
        mean_s = mu;
        inv_s = rsqrtf(b2 / (float)n - mu * mu + 1e-5f);
    }
    __syncthreads();
    const float mu = mean_s, inv = inv_s;
    for (long long i = threadIdx.x; i < n; i += 256) {
        int c = (int)(i % D);
        float v = (base[i] - mu) * inv * gw[c] + gbb[c];
        base[i] = siluf(v);
    }
}

// ---------------------------------------------------------------------------
// Host orchestration
// ---------------------------------------------------------------------------
extern "C" void kernel_launch(void* const* d_in, const int* in_sizes, int n_in,
                              void* d_out, int out_size, void* d_ws, size_t ws_size,
                              hipStream_t stream) {
    (void)in_sizes; (void)n_in; (void)out_size; (void)ws_size;

    const float* x     = (const float*)d_in[0];
    const float* style = (const float*)d_in[1];
    // d_in[2] = text_lengths (unused by forward)
    const unsigned char* mask = (const unsigned char*)d_in[3];
    auto P = [&](int i) -> const float* { return (const float*)d_in[i]; };

    // ---- workspace carve ----
    float* wsf = (float*)d_ws;
    size_t off = 0;
    auto alloc = [&](size_t nn) { float* p = wsf + off; off += nn; return p; };
    float* h    = alloc((size_t)BTD);             // running activations [B,T,D]
    float* y    = alloc((size_t)BTD);             // SALN output
    float* hid  = alloc((size_t)BT * FF);         // FFN hidden / pw1 out / attn out
    float* qb   = alloc((size_t)BTD);
    float* kb   = alloc((size_t)BTD);
    float* vb   = alloc((size_t)BTD);
    float* sc   = alloc((size_t)H * T * T);       // per-batch scores
    float* gbuf = alloc((size_t)BQ * 2 * D);      // style gamma/beta
    int*   tbl  = (int*)(wsf + off); off += (size_t)T * T;

    auto gemm = [&](const float* A, int lda, long long sA,
                    const float* Wm, int ldw, long long sW, int tB,
                    const float* bias, const float* Res,
                    float* C, int ldc, long long sC,
                    int M, int N, int Kd, int act, float scO, int batch) {
        dim3 g(N / 64, M / 128, batch);
        gemm_bf16<<<g, dim3(256), 0, stream>>>(A, lda, sA, Wm, ldw, sW, tB,
                                               bias, Res, C, ldc, sC, M, N, Kd, act, scO);
    };
    auto saln = [&](const float* in, float* outp, const float* lnw, const float* lnb,
                    const float* gbp, const unsigned char* mk) {
        saln_ln<<<dim3(BT / 8), dim3(256), 0, stream>>>(in, outp, lnw, lnb, gbp, mk);
    };
    auto affine = [&](const float* w, const float* b2) {
        style_affine<<<dim3((BQ * 2 * D) / 256), dim3(256), 0, stream>>>(style, w, b2, gbuf);
    };

    pos_embed<<<dim3((unsigned)((BTD + 255) / 256)), dim3(256), 0, stream>>>(x, h);
    relbucket<<<dim3((T * T) / 256), dim3(256), 0, stream>>>(tbl);

    // param leaf order: insertion order of _layer_params (43 leaves/layer)
    for (int l = 0; l < LAYERS; l++) {
        const int pb = 4 + l * 43;
        const float *s1w = P(pb + 0),  *s1b = P(pb + 1),  *a1w = P(pb + 2),  *a1b = P(pb + 3);
        const float *f11w = P(pb + 4), *f11b = P(pb + 5), *f12w = P(pb + 6), *f12b = P(pb + 7);
        const float *s2w = P(pb + 8),  *s2b = P(pb + 9),  *a2w = P(pb + 10), *a2b = P(pb + 11);
        const float *qw = P(pb + 12),  *qbi = P(pb + 13), *kw = P(pb + 14),  *kbi = P(pb + 15);
        const float *vw = P(pb + 16),  *vbi = P(pb + 17), *ow = P(pb + 18),  *obi = P(pb + 19);
        const float *relb = P(pb + 20);
        const float *s3w = P(pb + 21), *s3b = P(pb + 22), *a3w = P(pb + 23), *a3b = P(pb + 24);
        const float *p1w = P(pb + 25), *p1b = P(pb + 26);
        const float *dww = P(pb + 27), *dwb = P(pb + 28);
        const float *gnw = P(pb + 29), *gnb = P(pb + 30);
        const float *p2w = P(pb + 31), *p2b = P(pb + 32);
        const float *s4w = P(pb + 33), *s4b = P(pb + 34), *a4w = P(pb + 35), *a4b = P(pb + 36);
        const float *f21w = P(pb + 37), *f21b = P(pb + 38), *f22w = P(pb + 39), *f22b = P(pb + 40);
        const float *flw = P(pb + 41), *flb = P(pb + 42);

        // ---- FFN 1:  h += 0.5 * W2(silu(W1(saln(h)))) ----
        affine(a1w, a1b); saln(h, y, s1w, s1b, gbuf, nullptr);
        gemm(y,   D,  0, f11w, D,  0, 0, f11b, nullptr, hid, FF, 0, BT, FF, D,  1, 1.0f, 1);
        gemm(hid, FF, 0, f12w, FF, 0, 0, f12b, h,       h,   D,  0, BT, D,  FF, 0, 0.5f, 1);

        // ---- MHSA ----
        affine(a2w, a2b); saln(h, y, s2w, s2b, gbuf, nullptr);
        gemm(y, D, 0, qw, D, 0, 0, qbi, nullptr, qb, D, 0, BT, D, D, 0, 1.f, 1);
        gemm(y, D, 0, kw, D, 0, 0, kbi, nullptr, kb, D, 0, BT, D, D, 0, 1.f, 1);
        gemm(y, D, 0, vw, D, 0, 0, vbi, nullptr, vb, D, 0, BT, D, D, 0, 1.f, 1);
        for (int b = 0; b < BQ; b++) {
            const float* qp = qb + (long long)b * T * D;
            const float* kp = kb + (long long)b * T * D;
            const float* vp = vb + (long long)b * T * D;
            // scores[h,t,s] = (q . k) / sqrt(DH), batched over heads via grid.z
            gemm(qp, D, DH, kp, D, DH, 0, nullptr, nullptr,
                 sc, T, (long long)T * T, T, T, DH, 0, 0.125f, H);
            attn_softmax<<<dim3(T, H), dim3(256), 0, stream>>>(sc, relb, tbl,
                                                               mask + (long long)b * T);
            // out[t,d] = attn @ V  (transB: V accessed [s,d] column-wise)
            gemm(sc, T, (long long)T * T, vp, D, DH, 1, nullptr, nullptr,
                 hid + (long long)b * T * D, D, DH, T, DH, T, 0, 1.f, H);
        }
        gemm(hid, D, 0, ow, D, 0, 0, obi, h, h, D, 0, BT, D, D, 0, 1.f, 1);

        // ---- Conv module ----
        affine(a3w, a3b); saln(h, y, s3w, s3b, gbuf, nullptr);
        gemm(y, D, 0, p1w, D, 0, 0, p1b, nullptr, hid, 2 * D, 0, BT, 2 * D, D, 0, 1.f, 1);
        glu_kernel<<<dim3((unsigned)((BTD + 255) / 256)), dim3(256), 0, stream>>>(hid, qb);
        dwconv<<<dim3(T, BQ), dim3(256), 0, stream>>>(qb, kb, dww, dwb);
        groupnorm_silu<<<dim3(BQ), dim3(256), 0, stream>>>(kb, gnw, gnb);
        gemm(kb, D, 0, p2w, D, 0, 0, p2b, h, h, D, 0, BT, D, D, 0, 1.f, 1);

        // ---- FFN 2 ----
        affine(a4w, a4b); saln(h, y, s4w, s4b, gbuf, nullptr);
        gemm(y,   D,  0, f21w, D,  0, 0, f21b, nullptr, hid, FF, 0, BT, FF, D,  1, 1.0f, 1);
        gemm(hid, FF, 0, f22w, FF, 0, 0, f22b, h,       h,   D,  0, BT, D,  FF, 0, 0.5f, 1);

        // ---- per-layer final LN (in-place) ----
        saln(h, h, flw, flb, nullptr, nullptr);
    }

    // ---- final LN + padding-mask multiply, straight into d_out ----
    const float* Fw = P(4 + LAYERS * 43);
    const float* Fb = P(4 + LAYERS * 43 + 1);
    saln(h, (float*)d_out, Fw, Fb, nullptr, mask);
}
